// Model_GAT_12146167513766
// MI455X (gfx1250) — matrix operand order repaired and checked by reading, not compiled, not run on hardware
//
#include <hip/hip_runtime.h>

// ---------------- CDNA5 WMMA types & helpers ----------------
typedef __attribute__((ext_vector_type(16))) __bf16 v16bf;
typedef __attribute__((ext_vector_type(8)))  float  v8f;

constexpr int Bq = 4, INq = 768, Eq = 128, Cq = 256, PREq = 256, EDq = 32, VOCABq = 512;
constexpr int PIX = Eq * Eq;            // 16384 pixels per image
constexpr int CH1 = 2 * Cq + PREq;      // 768 conv1 input channels
constexpr int K2 = 9 * Cq;              // 2304 conv2 implicit-GEMM K

__device__ __forceinline__ float lrelu(float v) { return v > 0.f ? v : 0.1f * v; }

__device__ __forceinline__ v8f wmma_bf16(v16bf a, v16bf b, v8f c) {
  // D = A(16x32 bf16) * B(32x16 bf16) + C(16x16 f32)
  return __builtin_amdgcn_wmma_f32_16x16x32_bf16(false, a, false, b, (short)0, c, false, false);
}

// A fragment (16x32, row-major rows=M, K contiguous): src = base + row0*ld + k0
// Contiguous k order used here (A and B match): lane<16 -> K=0..15, lane>=16 -> K=16..31 (one contiguous run)
__device__ __forceinline__ v16bf ld_frag_a(const __bf16* src, int ld) {
  int lane = threadIdx.x & 31, m = lane & 15, hf = lane >> 4;
  const __bf16* p = src + m * ld + hf * 16;
  v16bf v;
#pragma unroll
  for (int j = 0; j < 16; ++j) v[j] = p[j];
  return v;
}

// B fragment (32x16, element (k,n) at src0 + n*ld + k, K contiguous): src = base + col0*ld + k0
// Contiguous k order used here (A and B match): lane<16 -> K=0..15, lane>=16 -> K=16..31 (one contiguous run)
__device__ __forceinline__ v16bf ld_frag_b(const __bf16* src, int ld) {
  int lane = threadIdx.x & 31, n = lane & 15, hf = lane >> 4;
  const __bf16* p = src + n * ld + hf * 16;
  v16bf v;
#pragma unroll
  for (int j = 0; j < 16; ++j) v[j] = p[j];
  return v;
}

__device__ __forceinline__ float wave_max(float v) {
#pragma unroll
  for (int s = 16; s > 0; s >>= 1) v = fmaxf(v, __shfl_xor(v, s, 32));
  return v;
}
__device__ __forceinline__ float wave_sum(float v) {
#pragma unroll
  for (int s = 16; s > 0; s >>= 1) v += __shfl_xor(v, s, 32);
  return v;
}

// ---------------- prep: vocab scalars + weight bf16 conversions ----------------
__global__ void prep_consts(const float* pw, const float* ew, const float* path_table,
                            const float* edge_table, const float* c1w, const float* xw1,
                            const float* yw1, const float* c2w, const float* xlw, const float* ylw,
                            float* pl, float* el, __bf16* wbf, __bf16* xw1bf, __bf16* yw1bf,
                            __bf16* a2, __bf16* xlwbf, __bf16* ylwbf) {
  const long long total = 512 + 3LL * 196608 + 3LL * 589824;
  for (long long t = (long long)blockIdx.x * blockDim.x + threadIdx.x; t < total;
       t += (long long)gridDim.x * blockDim.x) {
    long long u = t;
    if (u < 512) {
      float sp = 0.f, se = 0.f;
      for (int d = 0; d < EDq; ++d) {
        sp += path_table[u * EDq + d] * pw[d];
        se += edge_table[u * EDq + d] * ew[d];
      }
      pl[u] = lrelu(sp); el[u] = lrelu(se); continue;
    }
    u -= 512;
    if (u < 196608) { wbf[u] = (__bf16)c1w[u]; continue; }
    u -= 196608;
    if (u < 196608) { xw1bf[u] = (__bf16)xw1[u]; continue; }
    u -= 196608;
    if (u < 196608) { yw1bf[u] = (__bf16)yw1[u]; continue; }
    u -= 196608;
    if (u < 589824) {  // a2[oc][tap][ic] = c2w[oc][ic][dy][dx]  (K-contiguous per row)
      int ic = (int)(u & 255); int tap = (int)((u >> 8) % 9); int o = (int)(u / (9 * 256));
      int dy = tap / 3, dx = tap % 3;
      a2[u] = (__bf16)c2w[((o * 256 + ic) * 3 + dy) * 3 + dx]; continue;
    }
    u -= 589824;
    if (u < 589824) { xlwbf[u] = (__bf16)xlw[u]; continue; }
    u -= 589824;
    ylwbf[u] = (__bf16)ylw[u];
  }
}

// ---------------- prep: x/y [b][c][e] f32 -> xT/yT [b][e][c] bf16 ----------------
__global__ void prep_transpose_xy(const float* x, const float* y, __bf16* xT, __bf16* yT) {
  __shared__ float tile[32][33];
  int b = blockIdx.y, which = blockIdx.z;
  const float* src = which ? y : x;
  __bf16* dst = which ? yT : xT;
  int c0 = (blockIdx.x >> 2) * 32, e0 = (blockIdx.x & 3) * 32;
  int tx = threadIdx.x & 31, ty = threadIdx.x >> 5;
#pragma unroll
  for (int it = 0; it < 4; ++it) {
    int c = c0 + ty + it * 8;
    tile[ty + it * 8][tx] = src[(b * INq + c) * Eq + e0 + tx];
  }
  __syncthreads();
#pragma unroll
  for (int it = 0; it < 4; ++it) {
    int e = e0 + ty + it * 8;
    dst[(b * Eq + e) * INq + c0 + tx] = (__bf16)tile[tx][ty + it * 8];
  }
}

// ---------------- prep: pre_out [b][p][i][e] f32 -> h2[b][i][e][512+p] bf16 ----------------
__global__ void prep_preout(const float* pre, __bf16* h2) {
  __shared__ float tile[32][33];
  int b = blockIdx.z, i = blockIdx.y;
  int p0 = (blockIdx.x >> 2) * 32, e0 = (blockIdx.x & 3) * 32;
  int tx = threadIdx.x & 31, ty = threadIdx.x >> 5;
#pragma unroll
  for (int it = 0; it < 4; ++it) {
    int p = p0 + ty + it * 8;
    tile[ty + it * 8][tx] = pre[(((long long)b * PREq + p) * Eq + i) * Eq + e0 + tx];
  }
  __syncthreads();
#pragma unroll
  for (int it = 0; it < 4; ++it) {
    int e = e0 + ty + it * 8;
    h2[(((long long)b * Eq + i) * Eq + e) * CH1 + 512 + p0 + tx] = (__bf16)tile[tx][ty + it * 8];
  }
}

// ---------------- projections: x_f = xw1 @ x[b]  (M=256,N=128,K=768) ----------------
__global__ __launch_bounds__(512) void proj_gemm(const __bf16* xw1bf, const __bf16* yw1bf,
                                                 const __bf16* xT, const __bf16* yT, float* x_f,
                                                 float* y_f, __bf16* xfbf, __bf16* yfbf) {
  int b = blockIdx.x, which = blockIdx.y;
  const __bf16* A = which ? yw1bf : xw1bf;                 // [o][c]
  const __bf16* Bm = (which ? yT : xT) + b * Eq * INq;     // [e][c]
  float*  Of = (which ? y_f : x_f) + b * Cq * Eq;
  __bf16* Ob = (which ? yfbf : xfbf) + b * Cq * Eq;
  int w = threadIdx.x >> 5, lane = threadIdx.x & 31;
  int wm = (w & 3) * 64, wn = (w >> 2) * 32;
  v8f acc[4][2] = {};
  for (int k0 = 0; k0 < INq; k0 += 32) {
    v16bf bf_[2];
#pragma unroll
    for (int nt = 0; nt < 2; ++nt) bf_[nt] = ld_frag_b(Bm + (wn + nt * 16) * INq + k0, INq);
#pragma unroll
    for (int mt = 0; mt < 4; ++mt) {
      v16bf af = ld_frag_a(A + (wm + mt * 16) * INq + k0, INq);
#pragma unroll
      for (int nt = 0; nt < 2; ++nt) acc[mt][nt] = wmma_bf16(af, bf_[nt], acc[mt][nt]);
    }
  }
  int nl = lane & 15, hf = lane >> 4;
#pragma unroll
  for (int mt = 0; mt < 4; ++mt)
#pragma unroll
    for (int nt = 0; nt < 2; ++nt)
#pragma unroll
      for (int r = 0; r < 8; ++r) {
        int o = wm + mt * 16 + r + 8 * hf, e = wn + nt * 16 + nl;
        float v = acc[mt][nt][r];
        Of[o * Eq + e] = v;
        Ob[o * Eq + e] = (__bf16)v;
      }
}

// ---------------- x_self / y_other / logist ----------------
__global__ void self_logist(const float* x_f, const float* y_f, const float* xw2,
                            const float* xb2, const float* yw2, const float* yb2, float* logist) {
  __shared__ float xs[Eq], yo[Eq];
  int b = blockIdx.x, e = threadIdx.x;
  float sx = 0.f, sy = 0.f;
  for (int c = 0; c < Cq; ++c) {
    sx += xw2[c] * x_f[(b * Cq + c) * Eq + e];
    sy += yw2[c] * y_f[(b * Cq + c) * Eq + e];
  }
  xs[e] = sx + xb2[0];
  yo[e] = sy + yb2[0];
  __syncthreads();
  for (int idx = threadIdx.x; idx < PIX; idx += Eq) {
    int m = idx >> 7, n = idx & 127;
    logist[b * PIX + idx] = lrelu(xs[n] + yo[m]);
  }
}

// ---------------- per-entity biased attention + WMMA GEMMs -> h2[...,0..512) ----------------
__global__ __launch_bounds__(512) void attention_kernel(const float* logist, const float* pl,
                                                        const int* path_mat, const float* p_bias,
                                                        const __bf16* xfbf, const __bf16* yfbf,
                                                        const float* x_f, const float* y_f,
                                                        __bf16* h2) {
  __shared__ float  loc[Eq][Eq + 1];
  __shared__ __bf16 attn[Eq][Eq];
  __shared__ __bf16 attnT[Eq][Eq];
  int i = blockIdx.x, b = blockIdx.y;
  long long base = ((long long)b * Eq + i) * PIX;
  for (int idx = threadIdx.x; idx < PIX; idx += 512) {
    int m = idx >> 7, n = idx & 127;
    loc[m][n] = logist[b * PIX + idx] + pl[path_mat[base + idx]] + p_bias[base + idx];
  }
  __syncthreads();
  int w = threadIdx.x >> 5, lane = threadIdx.x & 31;
  // softmax over n (rows) -> attn[m][n]
#pragma unroll
  for (int rr = 0; rr < 8; ++rr) {
    int m = w * 8 + rr;
    float v[4], mx = -1e30f;
#pragma unroll
    for (int q = 0; q < 4; ++q) { v[q] = loc[m][lane + 32 * q]; mx = fmaxf(mx, v[q]); }
    mx = wave_max(mx);
    float s = 0.f;
#pragma unroll
    for (int q = 0; q < 4; ++q) { v[q] = __expf(v[q] - mx); s += v[q]; }
    s = 1.f / wave_sum(s);
#pragma unroll
    for (int q = 0; q < 4; ++q) attn[m][lane + 32 * q] = (__bf16)(v[q] * s);
  }
  // softmax over m (cols) -> attnT[n][m]
#pragma unroll
  for (int cc = 0; cc < 8; ++cc) {
    int n = w * 8 + cc;
    float v[4], mx = -1e30f;
#pragma unroll
    for (int q = 0; q < 4; ++q) { v[q] = loc[lane + 32 * q][n]; mx = fmaxf(mx, v[q]); }
    mx = wave_max(mx);
    float s = 0.f;
#pragma unroll
    for (int q = 0; q < 4; ++q) { v[q] = __expf(v[q] - mx); s += v[q]; }
    s = 1.f / wave_sum(s);
#pragma unroll
    for (int q = 0; q < 4; ++q) attnT[n][lane + 32 * q] = (__bf16)(v[q] * s);
  }
  __syncthreads();
  int wm = (w & 3) * 32, wn = (w >> 2) * 64;
  int nl = lane & 15, hf = lane >> 4;
  const float* Xf = x_f + b * Cq * Eq;
  const float* Yf = y_f + b * Cq * Eq;
  __bf16* H = h2 + ((long long)b * Eq + i) * Eq * CH1;
  // GEMM 1: x_ret[m,c] = lrelu(attn @ y_f^T + x_f^T)   (M=128,N=256,K=128)
  {
    const __bf16* Bx = yfbf + b * Cq * Eq;  // [c][n], K=n contiguous
    v8f acc[2][4] = {};
    for (int k0 = 0; k0 < Eq; k0 += 32) {
      v16bf bf_[4];
#pragma unroll
      for (int nt = 0; nt < 4; ++nt) bf_[nt] = ld_frag_b(Bx + (wn + nt * 16) * Eq + k0, Eq);
#pragma unroll
      for (int mt = 0; mt < 2; ++mt) {
        v16bf af = ld_frag_a(&attn[wm + mt * 16][0] + k0, Eq);
#pragma unroll
        for (int nt = 0; nt < 4; ++nt) acc[mt][nt] = wmma_bf16(af, bf_[nt], acc[mt][nt]);
      }
    }
#pragma unroll
    for (int mt = 0; mt < 2; ++mt)
#pragma unroll
      for (int nt = 0; nt < 4; ++nt)
#pragma unroll
        for (int r = 0; r < 8; ++r) {
          int m = wm + mt * 16 + r + 8 * hf, c = wn + nt * 16 + nl;
          H[(long long)m * CH1 + c] = (__bf16)lrelu(acc[mt][nt][r] + Xf[c * Eq + m]);
        }
  }
  // GEMM 2: y_ret[n,c] = lrelu(attnT @ x_f^T + y_f^T)
  {
    const __bf16* By = xfbf + b * Cq * Eq;
    v8f acc[2][4] = {};
    for (int k0 = 0; k0 < Eq; k0 += 32) {
      v16bf bf_[4];
#pragma unroll
      for (int nt = 0; nt < 4; ++nt) bf_[nt] = ld_frag_b(By + (wn + nt * 16) * Eq + k0, Eq);
#pragma unroll
      for (int mt = 0; mt < 2; ++mt) {
        v16bf af = ld_frag_a(&attnT[wm + mt * 16][0] + k0, Eq);
#pragma unroll
        for (int nt = 0; nt < 4; ++nt) acc[mt][nt] = wmma_bf16(af, bf_[nt], acc[mt][nt]);
      }
    }
#pragma unroll
    for (int mt = 0; mt < 2; ++mt)
#pragma unroll
      for (int nt = 0; nt < 4; ++nt)
#pragma unroll
        for (int r = 0; r < 8; ++r) {
          int n = wm + mt * 16 + r + 8 * hf, c = wn + nt * 16 + nl;
          H[(long long)n * CH1 + Cq + c] = (__bf16)lrelu(acc[mt][nt][r] + Yf[c * Eq + n]);
        }
  }
}

// ---------------- conv1 1x1: g1[b][pix][oc] = lrelu(W @ h2 + b)  (M=256,N=16384,K=768) ----------------
__global__ __launch_bounds__(512) void conv1_gemm(const __bf16* wbf, const __bf16* h2,
                                                  const float* c1b, __bf16* g1) {
  int nblk = blockIdx.x, b = blockIdx.y, mhalf = blockIdx.z;
  int w = threadIdx.x >> 5, lane = threadIdx.x & 31;
  int wm = mhalf * 128 + (w & 3) * 32;
  int wn = nblk * 128 + (w >> 2) * 32;
  const __bf16* Bm = h2 + (long long)b * PIX * CH1;
  v8f acc[2][2] = {};
  for (int k0 = 0; k0 < CH1; k0 += 32) {
    __builtin_prefetch((const void*)(Bm + (long long)wn * CH1 + k0 + 256), 0, 1);
    v16bf bf_[2];
#pragma unroll
    for (int nt = 0; nt < 2; ++nt) bf_[nt] = ld_frag_b(Bm + (long long)(wn + nt * 16) * CH1 + k0, CH1);
#pragma unroll
    for (int mt = 0; mt < 2; ++mt) {
      v16bf af = ld_frag_a(wbf + (wm + mt * 16) * CH1 + k0, CH1);
#pragma unroll
      for (int nt = 0; nt < 2; ++nt) acc[mt][nt] = wmma_bf16(af, bf_[nt], acc[mt][nt]);
    }
  }
  int nl = lane & 15, hf = lane >> 4;
#pragma unroll
  for (int mt = 0; mt < 2; ++mt)
#pragma unroll
    for (int nt = 0; nt < 2; ++nt)
#pragma unroll
      for (int r = 0; r < 8; ++r) {
        int o = wm + mt * 16 + r + 8 * hf, pix = wn + nt * 16 + nl;
        g1[((long long)b * PIX + pix) * Cq + o] = (__bf16)lrelu(acc[mt][nt][r] + c1b[o]);
      }
}

// ---------------- conv2 3x3 SAME implicit GEMM (M=256,N=16384,K=2304) -> men2rel (f32, d_out) ----------------
__global__ __launch_bounds__(512) void conv2_gemm(const __bf16* a2, const __bf16* g1,
                                                  const float* c2b, float* men) {
  int yrow = blockIdx.x, b = blockIdx.y, mhalf = blockIdx.z;
  int w = threadIdx.x >> 5, lane = threadIdx.x & 31;
  int wm = mhalf * 128 + (w & 3) * 32;
  int wx = (w >> 2) * 32;
  const __bf16* G = g1 + (long long)b * PIX * Cq;
  v8f acc[2][2] = {};
  int nl = lane & 15, hf = lane >> 4;
  for (int k0 = 0; k0 < K2; k0 += 32) {
    int tap = k0 >> 8, ic0 = k0 & 255;
    int dy = tap / 3 - 1, dx = tap % 3 - 1;
    int yy = yrow + dy;
    __builtin_prefetch((const void*)(a2 + wm * K2 + k0 + 64), 0, 1);
    v16bf bf_[2];
#pragma unroll
    for (int nt = 0; nt < 2; ++nt) {
      int xx = wx + nt * 16 + nl + dx;
      v16bf v;
      if (yy >= 0 && yy < Eq && xx >= 0 && xx < Eq) {
        const __bf16* p = G + ((long long)yy * Eq + xx) * Cq + ic0 + hf * 16;
#pragma unroll
        for (int j = 0; j < 16; ++j) v[j] = p[j];
      } else {
#pragma unroll
        for (int j = 0; j < 16; ++j) v[j] = (__bf16)0.f;
      }
      bf_[nt] = v;
    }
#pragma unroll
    for (int mt = 0; mt < 2; ++mt) {
      v16bf af = ld_frag_a(a2 + (wm + mt * 16) * K2 + k0, K2);
#pragma unroll
      for (int nt = 0; nt < 2; ++nt) acc[mt][nt] = wmma_bf16(af, bf_[nt], acc[mt][nt]);
    }
  }
#pragma unroll
  for (int mt = 0; mt < 2; ++mt)
#pragma unroll
    for (int nt = 0; nt < 2; ++nt)
#pragma unroll
      for (int r = 0; r < 8; ++r) {
        int o = wm + mt * 16 + r + 8 * hf;
        int pix = yrow * Eq + wx + nt * 16 + nl;
        men[((long long)b * Cq + o) * PIX + pix] = lrelu(acc[mt][nt][r] + c2b[o]);
      }
}

// ---------------- scores + gl ----------------
__global__ void scores_gl(const float* men, const float* score_w, const float* logist,
                          const float* el, const int* edge_mat, const float* m_bias, float* gl) {
  int b = blockIdx.y, hrow = blockIdx.x, wcol = threadIdx.x;
  float acc = 0.f;
  long long p = (long long)b * Cq * PIX + hrow * Eq + wcol;
  for (int c = 0; c < Cq; ++c) acc += score_w[c] * men[p + (long long)c * PIX];
  int idx = b * PIX + hrow * Eq + wcol;
  gl[idx] = logist[idx] + lrelu(acc) + el[edge_mat[idx]] + m_bias[idx];
}

// ---------------- global attention probabilities (row & column softmax, bf16) ----------------
__global__ __launch_bounds__(512) void prob_kernel(const float* gl, __bf16* Pbf, __bf16* Qbf) {
  __shared__ float g[Eq][Eq + 1];
  int b = blockIdx.x;
  for (int idx = threadIdx.x; idx < PIX; idx += 512) g[idx >> 7][idx & 127] = gl[b * PIX + idx];
  __syncthreads();
  int w = threadIdx.x >> 5, lane = threadIdx.x & 31;
#pragma unroll
  for (int rr = 0; rr < 8; ++rr) {
    int n = w * 8 + rr;
    {  // P[n][m] = softmax_m gl[n][m]
      float v[4], mx = -1e30f;
#pragma unroll
      for (int q = 0; q < 4; ++q) { v[q] = g[n][lane + 32 * q]; mx = fmaxf(mx, v[q]); }
      mx = wave_max(mx);
      float s = 0.f;
#pragma unroll
      for (int q = 0; q < 4; ++q) { v[q] = __expf(v[q] - mx); s += v[q]; }
      s = 1.f / wave_sum(s);
#pragma unroll
      for (int q = 0; q < 4; ++q) Pbf[(b * Eq + n) * Eq + lane + 32 * q] = (__bf16)(v[q] * s);
    }
    {  // Q[n][m] = softmax_m gl[m][n]
      float v[4], mx = -1e30f;
#pragma unroll
      for (int q = 0; q < 4; ++q) { v[q] = g[lane + 32 * q][n]; mx = fmaxf(mx, v[q]); }
      mx = wave_max(mx);
      float s = 0.f;
#pragma unroll
      for (int q = 0; q < 4; ++q) { v[q] = __expf(v[q] - mx); s += v[q]; }
      s = 1.f / wave_sum(s);
#pragma unroll
      for (int q = 0; q < 4; ++q) Qbf[(b * Eq + n) * Eq + lane + 32 * q] = (__bf16)(v[q] * s);
    }
  }
}

// ---------------- x_lin/y_lin: [e][c] @ [o][c]^T -> linT[b][o][e] bf16 (M=128,N=768,K=768) ----------------
__global__ __launch_bounds__(512) void lin_gemm(const __bf16* xT, const __bf16* yT,
                                                const __bf16* xlwbf, const __bf16* ylwbf,
                                                __bf16* xlinT, __bf16* ylinT) {
  int nblk = blockIdx.x, b = blockIdx.y, which = blockIdx.z;
  const __bf16* A  = (which ? yT : xT) + b * Eq * INq;     // [e][c]
  const __bf16* Bw = which ? ylwbf : xlwbf;                // [o][c]
  __bf16* O = (which ? ylinT : xlinT) + b * INq * Eq;      // [o][e]
  int w = threadIdx.x >> 5, lane = threadIdx.x & 31;
  int wm = (w & 3) * 32;
  int wn = nblk * 128 + (w >> 2) * 32;
  v8f acc[2][2] = {};
  for (int k0 = 0; k0 < INq; k0 += 32) {
    v16bf bf_[2];
#pragma unroll
    for (int nt = 0; nt < 2; ++nt) bf_[nt] = ld_frag_b(Bw + (wn + nt * 16) * INq + k0, INq);
#pragma unroll
    for (int mt = 0; mt < 2; ++mt) {
      v16bf af = ld_frag_a(A + (wm + mt * 16) * INq + k0, INq);
#pragma unroll
      for (int nt = 0; nt < 2; ++nt) acc[mt][nt] = wmma_bf16(af, bf_[nt], acc[mt][nt]);
    }
  }
  int nl = lane & 15, hf = lane >> 4;
#pragma unroll
  for (int mt = 0; mt < 2; ++mt)
#pragma unroll
    for (int nt = 0; nt < 2; ++nt)
#pragma unroll
      for (int r = 0; r < 8; ++r) {
        int e = wm + mt * 16 + r + 8 * hf, o = wn + nt * 16 + nl;
        O[o * Eq + e] = (__bf16)acc[mt][nt][r];
      }
}

// ---------------- final: out = P @ lin + residual  (M=128,N=768,K=128) ----------------
__global__ __launch_bounds__(512) void final_gemm(const __bf16* Pbf, const __bf16* Qbf,
                                                  const __bf16* xlinT, const __bf16* ylinT,
                                                  const float* x, const float* y, float* outx,
                                                  float* outy) {
  int nblk = blockIdx.x, b = blockIdx.y, which = blockIdx.z;
  const __bf16* A  = (which ? Qbf : Pbf) + b * PIX;          // [n][m]
  const __bf16* Bv = (which ? xlinT : ylinT) + b * INq * Eq; // [c][m]
  const float* res = (which ? y : x) + b * INq * Eq;         // [c][n]
  float* O = (which ? outy : outx) + b * INq * Eq;
  int w = threadIdx.x >> 5, lane = threadIdx.x & 31;
  int wm = (w & 3) * 32;
  int wn = nblk * 128 + (w >> 2) * 32;
  v8f acc[2][2] = {};
  for (int k0 = 0; k0 < Eq; k0 += 32) {
    v16bf bf_[2];
#pragma unroll
    for (int nt = 0; nt < 2; ++nt) bf_[nt] = ld_frag_b(Bv + (wn + nt * 16) * Eq + k0, Eq);
#pragma unroll
    for (int mt = 0; mt < 2; ++mt) {
      v16bf af = ld_frag_a(A + (wm + mt * 16) * Eq + k0, Eq);
#pragma unroll
      for (int nt = 0; nt < 2; ++nt) acc[mt][nt] = wmma_bf16(af, bf_[nt], acc[mt][nt]);
    }
  }
  int nl = lane & 15, hf = lane >> 4;
#pragma unroll
  for (int mt = 0; mt < 2; ++mt)
#pragma unroll
    for (int nt = 0; nt < 2; ++nt)
#pragma unroll
      for (int r = 0; r < 8; ++r) {
        int n = wm + mt * 16 + r + 8 * hf, c = wn + nt * 16 + nl;
        O[c * Eq + n] = acc[mt][nt][r] + res[c * Eq + n];
      }
}

// ---------------- host launcher ----------------
extern "C" void kernel_launch(void* const* d_in, const int* in_sizes, int n_in, void* d_out,
                              int out_size, void* d_ws, size_t ws_size, hipStream_t stream) {
  (void)in_sizes; (void)n_in; (void)out_size; (void)ws_size;
  const float* x       = (const float*)d_in[0];
  const float* y       = (const float*)d_in[1];
  const float* m_bias  = (const float*)d_in[2];
  const float* p_bias  = (const float*)d_in[3];
  const float* pre_out = (const float*)d_in[4];
  const float* xw1     = (const float*)d_in[5];
  const float* yw1     = (const float*)d_in[6];
  const float* xw2     = (const float*)d_in[7];
  const float* xb2     = (const float*)d_in[8];
  const float* yw2     = (const float*)d_in[9];
  const float* yb2     = (const float*)d_in[10];
  const float* pw      = (const float*)d_in[11];
  const float* ew      = (const float*)d_in[12];
  const float* c1w     = (const float*)d_in[13];
  const float* c1b     = (const float*)d_in[14];
  const float* c2w     = (const float*)d_in[15];
  const float* c2b     = (const float*)d_in[16];
  const float* score_w = (const float*)d_in[17];
  const float* xlw     = (const float*)d_in[18];
  const float* ylw     = (const float*)d_in[19];
  const float* ptab    = (const float*)d_in[20];
  const float* etab    = (const float*)d_in[21];
  const int*   edge_m  = (const int*)d_in[22];
  const int*   path_m  = (const int*)d_in[23];

  float* outx = (float*)d_out;
  float* outy = outx + (size_t)Bq * INq * Eq;
  float* men  = outx + (size_t)2 * Bq * INq * Eq;

  char* p = (char*)d_ws;
  auto alloc = [&](size_t bytes) -> void* {
    void* r = (void*)p;
    p += (bytes + 255) & ~(size_t)255;
    return r;
  };
  float*  pl     = (float*)alloc(512 * 4);
  float*  el     = (float*)alloc(512 * 4);
  float*  x_f    = (float*)alloc((size_t)Bq * Cq * Eq * 4);
  float*  y_f    = (float*)alloc((size_t)Bq * Cq * Eq * 4);
  __bf16* xfbf   = (__bf16*)alloc((size_t)Bq * Cq * Eq * 2);
  __bf16* yfbf   = (__bf16*)alloc((size_t)Bq * Cq * Eq * 2);
  float*  logist = (float*)alloc((size_t)Bq * PIX * 4);
  __bf16* wbf    = (__bf16*)alloc((size_t)Cq * CH1 * 2);
  __bf16* xw1bf  = (__bf16*)alloc((size_t)Cq * INq * 2);
  __bf16* yw1bf  = (__bf16*)alloc((size_t)Cq * INq * 2);
  __bf16* a2     = (__bf16*)alloc((size_t)Cq * K2 * 2);
  __bf16* xlwbf  = (__bf16*)alloc((size_t)INq * INq * 2);
  __bf16* ylwbf  = (__bf16*)alloc((size_t)INq * INq * 2);
  __bf16* xT     = (__bf16*)alloc((size_t)Bq * Eq * INq * 2);
  __bf16* yT     = (__bf16*)alloc((size_t)Bq * Eq * INq * 2);
  __bf16* h2     = (__bf16*)alloc((size_t)Bq * PIX * CH1 * 2);   // ~100.7 MB
  __bf16* g1     = (__bf16*)alloc((size_t)Bq * PIX * Cq * 2);    // ~33.6 MB
  float*  gl     = (float*)alloc((size_t)Bq * PIX * 4);
  __bf16* Pbf    = (__bf16*)alloc((size_t)Bq * PIX * 2);
  __bf16* Qbf    = (__bf16*)alloc((size_t)Bq * PIX * 2);
  __bf16* xlinT  = (__bf16*)alloc((size_t)Bq * INq * Eq * 2);
  __bf16* ylinT  = (__bf16*)alloc((size_t)Bq * INq * Eq * 2);

  prep_consts<<<2048, 256, 0, stream>>>(pw, ew, ptab, etab, c1w, xw1, yw1, c2w, xlw, ylw,
                                        pl, el, wbf, xw1bf, yw1bf, a2, xlwbf, ylwbf);
  prep_transpose_xy<<<dim3(96, Bq, 2), 256, 0, stream>>>(x, y, xT, yT);
  prep_preout<<<dim3(32, Eq, Bq), 256, 0, stream>>>(pre_out, h2);
  proj_gemm<<<dim3(Bq, 2), 512, 0, stream>>>(xw1bf, yw1bf, xT, yT, x_f, y_f, xfbf, yfbf);
  self_logist<<<Bq, Eq, 0, stream>>>(x_f, y_f, xw2, xb2, yw2, yb2, logist);
  attention_kernel<<<dim3(Eq, Bq), 512, 0, stream>>>(logist, pl, path_m, p_bias, xfbf, yfbf,
                                                     x_f, y_f, h2);
  conv1_gemm<<<dim3(128, Bq, 2), 512, 0, stream>>>(wbf, h2, c1b, g1);
  conv2_gemm<<<dim3(Eq, Bq, 2), 512, 0, stream>>>(a2, g1, c2b, men);
  scores_gl<<<dim3(Eq, Bq), Eq, 0, stream>>>(men, score_w, logist, el, edge_m, m_bias, gl);
  prob_kernel<<<Bq, 512, 0, stream>>>(gl, Pbf, Qbf);
  lin_gemm<<<dim3(6, Bq, 2), 512, 0, stream>>>(xT, yT, xlwbf, ylwbf, xlinT, ylinT);
  final_gemm<<<dim3(6, Bq, 2), 512, 0, stream>>>(Pbf, Qbf, xlinT, ylinT, x, y, outx, outy);
}
